// Conv2D_68444598829079
// MI455X (gfx1250) — matrix operand order repaired and checked by reading, not compiled, not run on hardware
//
#include <hip/hip_runtime.h>

#define IN_C  128
#define OUT_C 256
#define HH    56
#define WW    56
#define BBB   32
#define KSZ   1152            // IN_C * 9, reordered as k' = (kh*3+kw)*128 + c
#define HWPIX (HH * WW)       // 3136
#define NPIX  (BBB * HWPIX)   // 100352
#define LDA   72              // LDS row stride in bf16 (64 + 8 pad) -> conflict-free b128 reads

typedef __attribute__((ext_vector_type(16))) __bf16          v16bf;
typedef __attribute__((ext_vector_type(8)))  float           v8f;
typedef __attribute__((ext_vector_type(8)))  unsigned short  u16x8;
typedef __attribute__((ext_vector_type(16))) unsigned short  u16x16;
typedef __attribute__((ext_vector_type(4)))  unsigned short  u16x4;

__device__ __forceinline__ unsigned short f2bf(float f) {
  unsigned u = __builtin_bit_cast(unsigned, f);
  u += 0x7FFFu + ((u >> 16) & 1u);            // round-to-nearest-even
  return (unsigned short)(u >> 16);
}

__device__ __forceinline__ v16bf load_frag(const unsigned short* p, int second_off) {
  u16x8 lo = *(const u16x8*)p;                 // 16B-aligned ds_load_b128
  u16x8 hi = *(const u16x8*)(p + second_off);  // 16B-aligned ds_load_b128
  u16x16 t = __builtin_shufflevector(lo, hi, 0,1,2,3,4,5,6,7,8,9,10,11,12,13,14,15);
  return __builtin_bit_cast(v16bf, t);
}

// ---- one-shot: Wk (OIHW fp32) -> Wp[o][ (kh*3+kw)*128 + c ] bf16 ----------
__global__ __launch_bounds__(256)
void pack_weights(const float* __restrict__ Wk, unsigned short* __restrict__ Wp) {
  int idx = blockIdx.x * 256 + threadIdx.x;     // 0 .. OUT_C*KSZ-1
  if (idx < OUT_C * KSZ) {
    int o = idx / KSZ;
    int r = idx - o * KSZ;
    int p = r >> 7;                             // filter position 0..8
    int c = r & 127;                            // channel
    Wp[idx] = f2bf(Wk[o * KSZ + c * 9 + p]);    // coalesced writes, L2-hot reads
  }
}

// ---- implicit GEMM: M=256 (oc), N=100352 (pixels), K=1152 -----------------
__global__ __launch_bounds__(256)
void conv3x3_wmma_bf16(const float* __restrict__ X,
                       const unsigned short* __restrict__ Wp,
                       const float* __restrict__ Bias,
                       float* __restrict__ Out)
{
  __shared__ unsigned short Alds[128 * LDA];    // weight tile  [m][k]  (18.4 KB)
  __shared__ unsigned short Blds[256 * LDA];    // im2col tile  [n][k]  (36.9 KB)

  const int tid  = threadIdx.x;
  const int lane = tid & 31;
  const int wid  = tid >> 5;        // 0..7
  const int wm   = wid >> 2;        // 0..1 : 64-row M slab
  const int wn   = wid & 3;         // 0..3 : 64-col N slab
  const int lhi  = lane >> 4;       // 0/1
  const int llo  = lane & 15;

  const int m0 = blockIdx.y * 128;  // 0 or 128
  const int n0 = blockIdx.x * 256;  // pixel-tile base

  // fixed pixel owned by this thread for im2col staging (one pixel, 64 channels/step)
  const int np   = n0 + tid;                   // global output pixel
  const int ow   = np % WW;
  const int oh   = (np / WW) % HH;
  const int bimg = np / HWPIX;

  v8f acc[4][4] = {};

  for (int kb = 0; kb < KSZ; kb += 64) {
    const int fp  = kb >> 7;                   // filter position 0..8 (uniform!)
    const int ch0 = (kb >> 6) & 1 ? 64 : 0;    // channel half
    const int kh  = fp / 3, kw = fp - kh * 3;

    // ---- stage A: 128 rows x 64 k of pre-packed bf16 weights (pure b128 copy)
    const unsigned short* wsrc = Wp + (size_t)m0 * KSZ + kb;
    #pragma unroll
    for (int t = 0; t < 4; ++t) {
      int q   = tid + t * 256;                 // b128 id: 0..1023
      int row = q >> 3;
      int c8  = q & 7;
      u16x8 w = *(const u16x8*)(wsrc + (size_t)row * KSZ + c8 * 8);
      *(u16x8*)(&Alds[row * LDA + c8 * 8]) = w;
    }
    if (kb + 64 < KSZ)                          // prefetch next weight tile
      __builtin_prefetch(wsrc + 64 + (size_t)(tid & 127) * KSZ, 0, 0);

    // ---- stage B: constant-stride channel walk; bounds checked ONCE per step
    {
      const int  ih  = oh + kh - 1;
      const int  iw  = ow + kw - 1;
      const bool val = (unsigned)ih < (unsigned)HH && (unsigned)iw < (unsigned)WW;
      const float* xs = X + (((size_t)bimg * IN_C + ch0) * HH + ih) * WW + iw;
      unsigned short* bdst = &Blds[tid * LDA];
      #pragma unroll
      for (int kl = 0; kl < 64; kl += 4) {      // all loads: one base + imm offsets
        float v0 = val ? xs[(kl + 0) * HWPIX] : 0.0f;
        float v1 = val ? xs[(kl + 1) * HWPIX] : 0.0f;
        float v2 = val ? xs[(kl + 2) * HWPIX] : 0.0f;
        float v3 = val ? xs[(kl + 3) * HWPIX] : 0.0f;
        u16x4 h = { f2bf(v0), f2bf(v1), f2bf(v2), f2bf(v3) };
        *(u16x4*)(bdst + kl) = h;               // 8B-aligned ds_store_b64
      }
    }
    __syncthreads();

    // ---- compute: two K=32 slabs, 32 wmma per wave per K-step
    #pragma unroll
    for (int ks = 0; ks < 64; ks += 32) {
      v16bf a[4], b[4];
      #pragma unroll
      for (int i = 0; i < 4; ++i) {
        int row = wm * 64 + i * 16 + llo;       // A: M across lanes, K in lane
        a[i] = load_frag(&Alds[row * LDA + ks + lhi * 8], 16);
      }
      #pragma unroll
      for (int j = 0; j < 4; ++j) {
        int col = wn * 64 + j * 16 + llo;       // B: N across lanes, K in lane
        b[j] = load_frag(&Blds[col * LDA + ks + lhi * 8], 16);
      }
      #pragma unroll
      for (int i = 0; i < 4; ++i)
        #pragma unroll
        for (int j = 0; j < 4; ++j)
          acc[i][j] = __builtin_amdgcn_wmma_f32_16x16x32_bf16(
              false, a[i], false, b[j], (short)0, acc[i][j], false, false);
    }
    __syncthreads();
  }

  // ---- epilogue: C/D layout -> NCHW, add bias
  #pragma unroll
  for (int j = 0; j < 4; ++j) {
    int n  = n0 + wn * 64 + j * 16 + llo;       // global pixel (lane = N)
    int p  = n % HWPIX;
    int bi = n / HWPIX;
    float* obase = Out + (size_t)bi * OUT_C * HWPIX + p;
    #pragma unroll
    for (int i = 0; i < 4; ++i) {
      #pragma unroll
      for (int v = 0; v < 8; ++v) {
        int oc = m0 + wm * 64 + i * 16 + lhi * 8 + v;   // VGPR v = M row
        obase[(size_t)oc * HWPIX] = acc[i][j][v] + Bias[oc];
      }
    }
  }
}

extern "C" void kernel_launch(void* const* d_in, const int* in_sizes, int n_in,
                              void* d_out, int out_size, void* d_ws, size_t ws_size,
                              hipStream_t stream) {
  (void)in_sizes; (void)n_in; (void)ws_size; (void)out_size;
  const float* x    = (const float*)d_in[0];
  const float* Wk   = (const float*)d_in[1];
  const float* bias = (const float*)d_in[2];
  float*       out  = (float*)d_out;
  unsigned short* Wp = (unsigned short*)d_ws;   // 256*1152*2B = 576 KB scratch

  pack_weights<<<(OUT_C * KSZ + 255) / 256, 256, 0, stream>>>(Wk, Wp);

  dim3 grid(NPIX / 256, OUT_C / 128);           // (392, 2)
  conv3x3_wmma_bf16<<<grid, 256, 0, stream>>>(x, Wp, bias, out);
}
